// SPLICE_16003048145168
// MI455X (gfx1250) — matrix operand order repaired and checked by reading, not compiled, not run on hardware
//
#include <hip/hip_runtime.h>
#include <math.h>

// ---------------------------------------------------------------------------
// Types for CDNA5 WMMA (wave32, 16x16x32 bf16 -> f32)
// ---------------------------------------------------------------------------
typedef __bf16 bf16;
typedef __attribute__((ext_vector_type(16))) __bf16 v16bf;
typedef __attribute__((ext_vector_type(8)))  __bf16 bf16x8;
typedef __attribute__((ext_vector_type(4)))  __bf16 bf16x4;
typedef __attribute__((ext_vector_type(8)))  float  v8f;

// Problem constants (match reference setup_inputs)
constexpr int BN = 1024;    // batch rows
constexpr int CN = 10000;   // dictionary atoms
constexpr int DM = 512;     // embedding dim
constexpr float RHO     = 5.0f;
constexpr float INV_RHO = 1.0f / 5.0f;
constexpr float THR     = 0.01f / 5.0f;   // lam / rho
constexpr int N_ADMM = 100;
constexpr int N_NS   = 12;                // Newton-Schulz iterations for G^-1

// GEMM tiling
#define TMT 128
#define TNT 64
#define TKT 32
#define LDT 40   // padded LDS row stride (elements) -> 80B rows, 16B aligned

// ---------------------------------------------------------------------------
// helpers
// ---------------------------------------------------------------------------
__device__ inline float wsum(float v) {
#pragma unroll
  for (int o = 16; o > 0; o >>= 1) v += __shfl_xor(v, o, 32);
  return v;
}

__device__ inline void split1(float v, bf16& h, bf16& l) {
  h = (bf16)v;
  l = (bf16)(v - (float)h);
}

// A fragment: 16x32 bf16 tile, LDS row-major (stride LDT).
// lanes 0-15: rows m, K in {kh..kh+7, 16+kh..+7}, kh=0 ; lanes 16-31: kh=8
__device__ inline v16bf fragA(const bf16* __restrict__ base, int lane) {
  int m  = lane & 15;
  int kh = (lane >> 4) << 3;
  const bf16* p = base + m * LDT;
  bf16x8 a = *(const bf16x8*)(p + kh);
  bf16x8 b = *(const bf16x8*)(p + 16 + kh);
  return __builtin_shufflevector(a, b, 0,1,2,3,4,5,6,7,8,9,10,11,12,13,14,15);
}

// B fragment: 32x16 tile as LDS[n][k]. lanes 0-15: K in {0..7, 16..23}; lanes 16-31: {8..15, 24..31}
__device__ inline v16bf fragB(const bf16* __restrict__ base, int lane) {
  int n  = lane & 15;
  int kh = (lane >> 4) << 3;
  const bf16* p = base + n * LDT + kh;
  bf16x8 a = *(const bf16x8*)(p);
  bf16x8 b = *(const bf16x8*)(p + 16);
  return __builtin_shufflevector(a, b, 0,1,2,3,4,5,6,7,8,9,10,11,12,13,14,15);
}

// ---------------------------------------------------------------------------
// Generic f32-accurate GEMM via bf16x3 split WMMA.  C = A x B
// ABF=1: A supplied as pre-split bf16 pair (A0=hi, A1=lo), [M][K] row-major.
//        Staging = pure b128 copies.
// ABF=0: A0 = f32 [M][K]; staged with float4 loads + pack converts.
// BBF=1: B supplied as bf16 pair in [N][K] row-major (K-contiguous per n),
//        i.e. already in the LDS[n][k] orientation. Pure b128 copies.
// BBF=0: B0 = f32 [K][N] row-major; transpose-staged (NS GEMMs only).
// EPI=0: C0[idx] = v
// EPI=1: C0[idx] = v + (row==col)*RHO              (build G)
// EPI=2: ADMM: b=AUXh+AUXl, u=C1: x=(b-v)/rho; zn=max(x+u-thr,0);
//        C0=zn; C1=u+x-zn
// EPI=3: split v -> Ch[idx], Cl[idx]               (bf16-pair output)
// ---------------------------------------------------------------------------
template <int ABF, int BBF, int EPI>
__global__ __launch_bounds__(256) void k_gemm(
    const void* __restrict__ A0, const void* __restrict__ A1, int lda,
    const void* __restrict__ B0, const void* __restrict__ B1, int ldb,
    float* __restrict__ C0, float* __restrict__ C1,
    bf16* __restrict__ Ch, bf16* __restrict__ Cl,
    const bf16* __restrict__ AUXh, const bf16* __restrict__ AUXl,
    int M, int N, int K)
{
  __shared__ __align__(16) bf16 sAh[TMT * LDT];
  __shared__ __align__(16) bf16 sAl[TMT * LDT];
  __shared__ __align__(16) bf16 sBh[TNT * LDT];
  __shared__ __align__(16) bf16 sBl[TNT * LDT];

  const int tid  = threadIdx.x;
  const int lane = tid & 31;
  const int wave = tid >> 5;
  const int wm = wave & 3;    // 4 wave rows -> 32 rows each
  const int wn = wave >> 2;   // 2 wave cols -> 32 cols each
  const int m0 = blockIdx.y * TMT;
  const int n0 = blockIdx.x * TNT;

  v8f acc[2][2] = {};

  for (int k0 = 0; k0 < K; k0 += TKT) {
    __syncthreads();
    // ---- stage A tile (TMT x TKT) ----
    if (ABF) {
      const bf16* Ah = (const bf16*)A0;
      const bf16* Al = (const bf16*)A1;
#pragma unroll
      for (int g = tid; g < (TMT * TKT) / 8; g += 256) {   // 2 iters
        int r = g >> 2, c8 = (g & 3) << 3;
        int gm = m0 + r, gk = k0 + c8;
        bf16x8 vh = {}, vl = {};
        if (gm < M && gk + 8 <= K) {
          size_t off = (size_t)gm * lda + gk;
          vh = *(const bf16x8*)(Ah + off);
          vl = *(const bf16x8*)(Al + off);
        }
        *(bf16x8*)(sAh + r * LDT + c8) = vh;
        *(bf16x8*)(sAl + r * LDT + c8) = vl;
      }
    } else {
      const float* Af = (const float*)A0;
#pragma unroll
      for (int g = tid; g < (TMT * TKT) / 4; g += 256) {   // 4 iters
        int r = g >> 3, c4 = (g & 7) << 2;
        int gm = m0 + r, gk = k0 + c4;
        float4 v = make_float4(0.f, 0.f, 0.f, 0.f);
        if (gm < M && gk + 4 <= K)
          v = *(const float4*)(Af + (size_t)gm * lda + gk);
        float vv[4] = {v.x, v.y, v.z, v.w};
        bf16x4 h4, l4;
#pragma unroll
        for (int j = 0; j < 4; j++) {
          bf16 h, l;
          split1(vv[j], h, l);
          h4[j] = h;
          l4[j] = l;
        }
        *(bf16x4*)(sAh + r * LDT + c4) = h4;
        *(bf16x4*)(sAl + r * LDT + c4) = l4;
      }
    }
    // ---- stage B tile (TNT x TKT in [n][k] orientation) ----
    if (BBF) {
      const bf16* Bh_ = (const bf16*)B0;
      const bf16* Bl_ = (const bf16*)B1;
#pragma unroll
      for (int g = tid; g < (TNT * TKT) / 8; g += 256) {   // 1 iter
        int n = g >> 2, c8 = (g & 3) << 3;
        int gn = n0 + n, gk = k0 + c8;
        bf16x8 vh = {}, vl = {};
        if (gn < N && gk + 8 <= K) {
          size_t off = (size_t)gn * ldb + gk;
          vh = *(const bf16x8*)(Bh_ + off);
          vl = *(const bf16x8*)(Bl_ + off);
        }
        *(bf16x8*)(sBh + n * LDT + c8) = vh;
        *(bf16x8*)(sBl + n * LDT + c8) = vl;
      }
    } else {
      const float* Bf = (const float*)B0;
#pragma unroll
      for (int g = tid; g < (TKT * TNT) / 4; g += 256) {   // 2 iters
        int c = g >> 4, n4 = (g & 15) << 2;
        int gk = k0 + c;
        float4 v = make_float4(0.f, 0.f, 0.f, 0.f);
        if (gk < K && n0 + n4 + 4 <= N)
          v = *(const float4*)(Bf + (size_t)gk * ldb + n0 + n4);
        bf16 h, l;
        split1(v.x, h, l); sBh[(n4+0)*LDT + c] = h; sBl[(n4+0)*LDT + c] = l;
        split1(v.y, h, l); sBh[(n4+1)*LDT + c] = h; sBl[(n4+1)*LDT + c] = l;
        split1(v.z, h, l); sBh[(n4+2)*LDT + c] = h; sBl[(n4+2)*LDT + c] = l;
        split1(v.w, h, l); sBh[(n4+3)*LDT + c] = h; sBl[(n4+3)*LDT + c] = l;
      }
    }
    __syncthreads();

    // ---- compute: 2x2 subtiles of 16x16 per wave, bf16x3 split ----
    v16bf bh[2], bl[2];
#pragma unroll
    for (int cb = 0; cb < 2; cb++) {
      int nb = (wn * 32 + cb * 16) * LDT;
      bh[cb] = fragB(sBh + nb, lane);
      bl[cb] = fragB(sBl + nb, lane);
    }
#pragma unroll
    for (int rb = 0; rb < 2; rb++) {
      int mb = (wm * 32 + rb * 16) * LDT;
      v16bf ah = fragA(sAh + mb, lane);
      v16bf al = fragA(sAl + mb, lane);
#pragma unroll
      for (int cb = 0; cb < 2; cb++) {
        acc[rb][cb] = __builtin_amdgcn_wmma_f32_16x16x32_bf16(
            false, ah, false, bl[cb], (short)0, acc[rb][cb], false, false);
        acc[rb][cb] = __builtin_amdgcn_wmma_f32_16x16x32_bf16(
            false, al, false, bh[cb], (short)0, acc[rb][cb], false, false);
        acc[rb][cb] = __builtin_amdgcn_wmma_f32_16x16x32_bf16(
            false, ah, false, bh[cb], (short)0, acc[rb][cb], false, false);
      }
    }
  }

  // ---- epilogue (C/D layout: col n = lane&15, half mh = lane>>4) ----
  const int cn = lane & 15;
  const int mh = lane >> 4;
#pragma unroll
  for (int rb = 0; rb < 2; rb++) {
#pragma unroll
    for (int cb = 0; cb < 2; cb++) {
      int gcol = n0 + wn * 32 + cb * 16 + cn;
      if (gcol >= N) continue;
#pragma unroll
      for (int r = 0; r < 8; r++) {
        int grow = m0 + wm * 32 + rb * 16 + mh * 8 + r;
        if (grow >= M) continue;
        int idx = grow * N + gcol;
        float v = acc[rb][cb][r];
        if (EPI == 0) {
          C0[idx] = v;
        } else if (EPI == 1) {
          C0[idx] = v + ((grow == gcol) ? RHO : 0.0f);
        } else if (EPI == 2) {
          float bb = (float)AUXh[idx] + (float)AUXl[idx];
          float uo = C1[idx];
          float x  = (bb - v) * INV_RHO;
          float zn = fmaxf(x + uo - THR, 0.0f);
          C0[idx] = zn;
          C1[idx] = uo + x - zn;
        } else {
          bf16 h, l;
          split1(v, h, l);
          Ch[idx] = h;
          Cl[idx] = l;
        }
      }
    }
  }
}

// ---------------------------------------------------------------------------
// one-time split/transpose of dictionary: D[CN][DM] f32
//   -> Dh/Dl  [CN][DM] bf16 pair   (K-contiguous for B = D^T GEMMs)
//   -> DTh/DTl[DM][CN] bf16 pair   (K-contiguous for B = D GEMMs, A = D^T)
// ---------------------------------------------------------------------------
__global__ __launch_bounds__(256) void k_transplit(
    const float* __restrict__ D, bf16* __restrict__ Dh, bf16* __restrict__ Dl,
    bf16* __restrict__ DTh, bf16* __restrict__ DTl)
{
  __shared__ float tile[64][65];
  int c0 = blockIdx.y * 64;   // atom block
  int d0 = blockIdx.x * 64;   // dim block
  for (int e = threadIdx.x; e < 64 * 64; e += 256) {
    int r = e >> 6, j = e & 63;
    int gc = c0 + r, gd = d0 + j;
    float v = (gc < CN) ? D[(size_t)gc * DM + gd] : 0.0f;
    tile[r][j] = v;
    if (gc < CN) {
      bf16 h, l;
      split1(v, h, l);
      Dh[(size_t)gc * DM + gd] = h;
      Dl[(size_t)gc * DM + gd] = l;
    }
  }
  __syncthreads();
  for (int e = threadIdx.x; e < 64 * 64; e += 256) {
    int r = e >> 6, j = e & 63;
    int gd = d0 + r, gc = c0 + j;
    if (gc < CN) {
      bf16 h, l;
      split1(tile[j][r], h, l);
      DTh[(size_t)gd * CN + gc] = h;
      DTl[(size_t)gd * CN + gc] = l;
    }
  }
}

// ---------------------------------------------------------------------------
// elementwise / row kernels
// ---------------------------------------------------------------------------
__global__ void k_fill(float* __restrict__ p, int n, float v) {
  for (int i = blockIdx.x * blockDim.x + threadIdx.x; i < n;
       i += gridDim.x * blockDim.x)
    p[i] = v;
}

// b = Aty + rho*(z-u), Aty given as bf16 pair; output bf16 pair
__global__ void k_makeb(const bf16* __restrict__ ath, const bf16* __restrict__ atl,
                        const float* __restrict__ Z, const float* __restrict__ U,
                        bf16* __restrict__ bh, bf16* __restrict__ bl, int n8) {
  for (int i = blockIdx.x * blockDim.x + threadIdx.x; i < n8;
       i += gridDim.x * blockDim.x) {
    bf16x8 AH = ((const bf16x8*)ath)[i];
    bf16x8 AL = ((const bf16x8*)atl)[i];
    float zv[8], uv[8];
    *(float4*)&zv[0] = ((const float4*)Z)[2 * i];
    *(float4*)&zv[4] = ((const float4*)Z)[2 * i + 1];
    *(float4*)&uv[0] = ((const float4*)U)[2 * i];
    *(float4*)&uv[4] = ((const float4*)U)[2 * i + 1];
    bf16x8 oh, ol;
#pragma unroll
    for (int j = 0; j < 8; j++) {
      float v = (float)AH[j] + (float)AL[j] + RHO * (zv[j] - uv[j]);
      bf16 h, l;
      split1(v, h, l);
      oh[j] = h;
      ol[j] = l;
    }
    ((bf16x8*)bh)[i] = oh;
    ((bf16x8*)bl)[i] = ol;
  }
}

__global__ void k_ns_update(float* __restrict__ X, const float* __restrict__ T2, int n) {
  for (int i = blockIdx.x * blockDim.x + threadIdx.x; i < n;
       i += gridDim.x * blockDim.x)
    X[i] = 2.0f * X[i] - T2[i];
}

// max row-sum of |G| -> atomicMax (positive float bits monotone as uint)
__global__ void k_gnorm(const float* __restrict__ G, float* __restrict__ sc) {
  int row  = blockIdx.x * 8 + (threadIdx.x >> 5);
  int lane = threadIdx.x & 31;
  float s = 0.0f;
  for (int j = lane; j < DM; j += 32) s += fabsf(G[row * DM + j]);
  s = wsum(s);
  if (lane == 0) atomicMax((unsigned int*)sc, __float_as_uint(s));
}

__global__ void k_initX(float* __restrict__ X, const float* __restrict__ sc) {
  float inv = 1.0f / sc[0];
  for (int e = blockIdx.x * blockDim.x + threadIdx.x; e < DM * DM;
       e += gridDim.x * blockDim.x) {
    int i = e / DM, j = e % DM;
    X[e] = (i == j) ? inv : 0.0f;
  }
}

// split X (symmetric) into bf16 pair, straight layout
__global__ void k_splitX(const float* __restrict__ X, bf16* __restrict__ Xh,
                         bf16* __restrict__ Xl) {
  for (int e = blockIdx.x * blockDim.x + threadIdx.x; e < DM * DM;
       e += gridDim.x * blockDim.x) {
    bf16 h, l;
    split1(X[e], h, l);
    Xh[e] = h;
    Xl[e] = l;
  }
}

// Y = normalize(normalize(img) - mean), written as bf16 pair; one wave per row
__global__ void k_center(const float* __restrict__ img, const float* __restrict__ mean,
                         bf16* __restrict__ Yh, bf16* __restrict__ Yl) {
  int row  = blockIdx.x * 8 + (threadIdx.x >> 5);
  int lane = threadIdx.x & 31;
  float v[16];
  float ss = 0.0f;
#pragma unroll
  for (int j = 0; j < 16; j++) {
    v[j] = img[row * DM + lane + 32 * j];
    ss += v[j] * v[j];
  }
  ss = wsum(ss);
  float sc = 1.0f / fmaxf(sqrtf(ss), 1e-12f);
  float s2 = 0.0f;
#pragma unroll
  for (int j = 0; j < 16; j++) {
    v[j] = v[j] * sc - mean[lane + 32 * j];
    s2 += v[j] * v[j];
  }
  s2 = wsum(s2);
  float c2 = 1.0f / fmaxf(sqrtf(s2), 1e-12f);
#pragma unroll
  for (int j = 0; j < 16; j++) {
    bf16 h, l;
    split1(v[j] * c2, h, l);
    Yh[row * DM + lane + 32 * j] = h;
    Yl[row * DM + lane + 32 * j] = l;
  }
}

// dst = normalize(src), one wave per row
__global__ void k_norm(const float* __restrict__ src, float* __restrict__ dst) {
  int row  = blockIdx.x * 8 + (threadIdx.x >> 5);
  int lane = threadIdx.x & 31;
  float v[16];
  float ss = 0.0f;
#pragma unroll
  for (int j = 0; j < 16; j++) {
    v[j] = src[row * DM + lane + 32 * j];
    ss += v[j] * v[j];
  }
  ss = wsum(ss);
  float sc = 1.0f / fmaxf(sqrtf(ss), 1e-12f);
#pragma unroll
  for (int j = 0; j < 16; j++) dst[row * DM + lane + 32 * j] = v[j] * sc;
}

// out = normalize(normalize(rr) + mean)
__global__ void k_final(const float* __restrict__ rr, const float* __restrict__ mean,
                        float* __restrict__ out) {
  int row  = blockIdx.x * 8 + (threadIdx.x >> 5);
  int lane = threadIdx.x & 31;
  float v[16];
  float ss = 0.0f;
#pragma unroll
  for (int j = 0; j < 16; j++) {
    v[j] = rr[row * DM + lane + 32 * j];
    ss += v[j] * v[j];
  }
  ss = wsum(ss);
  float sc = 1.0f / fmaxf(sqrtf(ss), 1e-12f);
  float s2 = 0.0f;
#pragma unroll
  for (int j = 0; j < 16; j++) {
    v[j] = v[j] * sc + mean[lane + 32 * j];
    s2 += v[j] * v[j];
  }
  s2 = wsum(s2);
  float c2 = 1.0f / fmaxf(sqrtf(s2), 1e-12f);
#pragma unroll
  for (int j = 0; j < 16; j++) out[row * DM + lane + 32 * j] = v[j] * c2;
}

// ---------------------------------------------------------------------------
// driver
// ---------------------------------------------------------------------------
extern "C" void kernel_launch(void* const* d_in, const int* in_sizes, int n_in,
                              void* d_out, int out_size, void* d_ws, size_t ws_size,
                              hipStream_t stream) {
  (void)in_sizes; (void)n_in; (void)out_size; (void)ws_size;
  const float* img  = (const float*)d_in[0];
  const float* txt  = (const float*)d_in[1];
  const float* mean = (const float*)d_in[2];
  const float* Dct  = (const float*)d_in[3];   // [CN, DM]
  float* out = (float*)d_out;                  // recon [BN,DM] then txt [BN,DM]

  char* w = (char*)d_ws;
  auto alloc = [&](size_t bytes) -> void* {
    void* p = (void*)w;
    w += (bytes + 255) & ~size_t(255);
    return p;
  };
  const size_t BC = (size_t)BN * CN;
  // bf16 pairs
  bf16* Yh  = (bf16*)alloc((size_t)BN * DM * 2);
  bf16* Yl  = (bf16*)alloc((size_t)BN * DM * 2);
  bf16* Ath = (bf16*)alloc(BC * 2);
  bf16* Atl = (bf16*)alloc(BC * 2);
  bf16* bh  = (bf16*)alloc(BC * 2);
  bf16* bl  = (bf16*)alloc(BC * 2);
  bf16* th  = (bf16*)alloc((size_t)BN * DM * 2);
  bf16* tl  = (bf16*)alloc((size_t)BN * DM * 2);
  bf16* sh  = (bf16*)alloc((size_t)BN * DM * 2);
  bf16* sl  = (bf16*)alloc((size_t)BN * DM * 2);
  bf16* Dh  = (bf16*)alloc((size_t)CN * DM * 2);
  bf16* Dl  = (bf16*)alloc((size_t)CN * DM * 2);
  bf16* DTh = (bf16*)alloc((size_t)CN * DM * 2);
  bf16* DTl = (bf16*)alloc((size_t)CN * DM * 2);
  bf16* Xh  = (bf16*)alloc((size_t)DM * DM * 2);
  bf16* Xl  = (bf16*)alloc((size_t)DM * DM * 2);
  // f32 state
  float* Z  = (float*)alloc(BC * 4);
  float* U  = (float*)alloc(BC * 4);
  float* G  = (float*)alloc((size_t)DM * DM * 4);
  float* X  = (float*)alloc((size_t)DM * DM * 4);
  float* T1 = (float*)alloc((size_t)DM * DM * 4);
  float* T2 = (float*)alloc((size_t)DM * DM * 4);
  float* RR = (float*)alloc((size_t)BN * DM * 4);
  float* SC = (float*)alloc(256);

  dim3 blk(256);
  // encode inputs
  k_center<<<BN / 8, blk, 0, stream>>>(img, mean, Yh, Yl);
  k_norm<<<BN / 8, blk, 0, stream>>>(txt, out + (size_t)BN * DM);

  // dictionary split + transpose (one time)
  k_transplit<<<dim3(DM / 64, (CN + 63) / 64), blk, 0, stream>>>(Dct, Dh, Dl, DTh, DTl);

  // G = D^T D + rho I : A = D^T (pair, lda=CN), B needs LDS[n][k]=D[k][n]=DT pair
  dim3 gG(DM / TNT, DM / TMT);
  k_gemm<1, 1, 1><<<gG, blk, 0, stream>>>(DTh, DTl, CN, DTh, DTl, CN,
                                          G, nullptr, nullptr, nullptr,
                                          nullptr, nullptr, DM, DM, CN);

  // X ~= G^-1 via Newton-Schulz, X0 = I / ||G||_inf
  k_fill<<<1, 32, 0, stream>>>(SC, 1, 0.0f);
  k_gnorm<<<DM / 8, blk, 0, stream>>>(G, SC);
  k_initX<<<256, blk, 0, stream>>>(X, SC);
  for (int it = 0; it < N_NS; it++) {
    k_gemm<0, 0, 0><<<gG, blk, 0, stream>>>(G, nullptr, DM, X, nullptr, DM,
                                            T1, nullptr, nullptr, nullptr,
                                            nullptr, nullptr, DM, DM, DM);
    k_gemm<0, 0, 0><<<gG, blk, 0, stream>>>(X, nullptr, DM, T1, nullptr, DM,
                                            T2, nullptr, nullptr, nullptr,
                                            nullptr, nullptr, DM, DM, DM);
    k_ns_update<<<256, blk, 0, stream>>>(X, T2, DM * DM);
  }
  k_splitX<<<256, blk, 0, stream>>>(X, Xh, Xl);

  // Aty = Y @ D^T : B[k][n] = D[n][k] -> Dh/Dl direct (ldb=DM); pair output
  dim3 gWide((CN + TNT - 1) / TNT, BN / TMT);
  k_gemm<1, 1, 3><<<gWide, blk, 0, stream>>>(Yh, Yl, DM, Dh, Dl, DM,
                                             nullptr, nullptr, Ath, Atl,
                                             nullptr, nullptr, BN, CN, DM);
  k_fill<<<2048, blk, 0, stream>>>(Z, (int)BC, 0.0f);
  k_fill<<<2048, blk, 0, stream>>>(U, (int)BC, 0.0f);

  // ADMM iterations
  dim3 gTall(DM / TNT, BN / TMT);
  for (int it = 0; it < N_ADMM; it++) {
    k_makeb<<<2560, blk, 0, stream>>>(Ath, Atl, Z, U, bh, bl, (int)(BC / 8));
    // t = b @ D : B LDS[n][k] = D[k][n] = DT pair (ldb=CN); pair out
    k_gemm<1, 1, 3><<<gTall, blk, 0, stream>>>(bh, bl, CN, DTh, DTl, CN,
                                               nullptr, nullptr, th, tl,
                                               nullptr, nullptr, BN, DM, CN);
    // s = t @ Ginv : X symmetric -> Xh/Xl direct
    k_gemm<1, 1, 3><<<gTall, blk, 0, stream>>>(th, tl, DM, Xh, Xl, DM,
                                               nullptr, nullptr, sh, sl,
                                               nullptr, nullptr, BN, DM, DM);
    // val = s @ D^T ; fused ADMM update of z,u (reads b pair, u)
    k_gemm<1, 1, 2><<<gWide, blk, 0, stream>>>(sh, sl, DM, Dh, Dl, DM,
                                               Z, U, nullptr, nullptr,
                                               bh, bl, BN, CN, DM);
  }

  // recon = normalize(normalize(z @ D) + mean)
  k_gemm<0, 1, 0><<<gTall, blk, 0, stream>>>(Z, nullptr, CN, DTh, DTl, CN,
                                             RR, nullptr, nullptr, nullptr,
                                             nullptr, nullptr, BN, DM, CN);
  k_final<<<BN / 8, blk, 0, stream>>>(RR, mean, out);
}
